// SwinTransformerBlock_83339545412158
// MI455X (gfx1250) — matrix operand order repaired and checked by reading, not compiled, not run on hardware
//
#include <hip/hip_runtime.h>
#include <hip/hip_bf16.h>

// ---------------------------------------------------------------------------
// SwinV2 block for MI455X (gfx1250, wave32, WMMA f32_16x16x32_f16)
// ws requirement: ~152 MB (padded f16 QKV cache + transposed f16 weights)
// d_out doubles as the post-attention intermediate Y (fully rewritten each call)
// - All WMMA B-operands are N-major (transposed): fragment = one 32B vector load
// - QKV cache padded to 64 tokens/window: k1 epilogue stores are unconditional
// - V cached dd-major so k2 stages Q/K/V with one v8h copy per thread each
// ---------------------------------------------------------------------------

typedef __attribute__((ext_vector_type(16))) _Float16 v16h;
typedef __attribute__((ext_vector_type(8)))  _Float16 v8h;
typedef __attribute__((ext_vector_type(8)))  float    v8f;

#define WIN     7
#define N_TOK   49
#define C_DIM   96
#define HEADS   3
#define DHEAD   32
#define NWIN    64
#define BATCH   64
#define HW_     56
#define SHIFT_  3
#define B_WIN   (BATCH*NWIN)      /* 4096 windows */
#define HIDDEN  384
#define METAD   128

// workspace layout (bytes)
#define BIAS_OFF  0                                /* 3*49*49 f32 = 28812 B  */
#define PART_OFF  (32*1024)                        /* 4096*2 f32  = 32768 B  */
#define SUMS_OFF  (PART_OFF + 4096*2*4)            /* 2 f32                  */
#define QWT_OFF   (80*1024)                        /* 288x96  f16 = 55296 B  */
#define PWT_OFF   (140*1024)                       /* 96x96   f16 = 18432 B  */
#define F1T_OFF   (164*1024)                       /* 384x96  f16 = 73728 B  */
#define F2T_OFF   (244*1024)                       /* 96x384  f16 = 73728 B  */
#define QKV_OFF   (384*1024)
/* padded: 64 tokens per (window,head); Q/K token-major, V dd-major */
#define QKV_ELEMS ((size_t)B_WIN*HEADS*64*DHEAD)   /* 25,165,824 halfs each  */

__device__ __forceinline__ v8f wmma16(v16h a, v16h b, v8f c) {
    return __builtin_amdgcn_wmma_f32_16x16x32_f16(
        /*neg_a=*/false, a, /*neg_b=*/false, b,
        /*c_mod=*/(short)0, c, /*reuse_a=*/false, /*reuse_b=*/false);
}

__device__ __forceinline__ float geluf(float v) {
    const float c = 0.7978845608028654f;
    float t = tanhf(c * (v + 0.044715f * v * v * v));
    return 0.5f * v * (1.0f + t);
}

// A fragment: row-major f16 matrix (M x lda), tile A(16x32) at (m0,k0).
// lanes 0-15: halves i -> k0+i; lanes 16-31: +16.
__device__ __forceinline__ v16h ldA(const _Float16* base, int lda, int m0, int k0) {
    int lane = threadIdx.x & 31;
    int m  = m0 + (lane & 15);
    int kb = k0 + ((lane >> 4) << 4);
    const _Float16* p = base + (size_t)m * lda + kb;
    v8h lo = *(const v8h*)(p);        // k = kb .. kb+7
    v8h hi = *(const v8h*)(p + 8);    // k = kb+8 .. kb+15
    v16h r;
#pragma unroll
    for (int i = 0; i < 8; ++i) { r[i] = lo[i]; r[8 + i] = hi[i]; }
    return r;
}

// B fragment from an N-major (transposed) f16 matrix: base[n*ldb + k].
// N = lane&15; lanes 0-15 hold K=k0..k0+15, lanes 16-31 hold K=k0+16..k0+31.
__device__ __forceinline__ v16h ldBT(const _Float16* base, int ldb, int k0, int n0) {
    int lane = threadIdx.x & 31;
    int n  = n0 + (lane & 15);
    int kb = k0 + ((lane >> 4) << 4);
    return *(const v16h*)(base + (size_t)n * ldb + kb);
}

__device__ __forceinline__ int region56(int g) { return g < 49 ? 0 : (g < 53 ? 1 : 2); }

// ---------------------------------------------------------------------------
// k0: log-CPB bias table  bias[h][i][j], h<3, i,j<49
// ---------------------------------------------------------------------------
__global__ void k0_bias(const float* __restrict__ w1, const float* __restrict__ b1,
                        const float* __restrict__ w2, const float* __restrict__ b2,
                        char* __restrict__ ws) {
    int p = blockIdx.x * 256 + threadIdx.x;
    if (p >= N_TOK * N_TOK) return;
    int i = p / N_TOK, j = p % N_TOK;
    float dy = (float)(i / WIN - j / WIN);
    float dx = (float)(i % WIN - j % WIN);
    float ry = copysignf(log1pf(fabsf(dy)), dy);
    float rx = copysignf(log1pf(fabsf(dx)), dx);
    float o0 = 0.f, o1 = 0.f, o2 = 0.f;
    for (int t = 0; t < METAD; ++t) {
        float hv = geluf(ry * w1[t] + rx * w1[METAD + t] + b1[t]);
        o0 += hv * w2[t * 3 + 0];
        o1 += hv * w2[t * 3 + 1];
        o2 += hv * w2[t * 3 + 2];
    }
    float* bias = (float*)(ws + BIAS_OFF);
    bias[0 * 2401 + p] = o0 + b2[0];
    bias[1 * 2401 + p] = o1 + b2[1];
    bias[2 * 2401 + p] = o2 + b2[2];
}

// ---------------------------------------------------------------------------
// k0w: transpose + f16-convert all GEMM weights into N-major layout in ws
// ---------------------------------------------------------------------------
__global__ void k0w_transpose(const float* __restrict__ qkv_w,
                              const float* __restrict__ proj_w,
                              const float* __restrict__ f1w,
                              const float* __restrict__ f2w,
                              char* __restrict__ ws) {
    int idx = blockIdx.x * 256 + threadIdx.x;
    if (idx < 288 * 96) {                         // QWT[n<288][k<96]
        int n = idx / 96, k = idx % 96;
        ((_Float16*)(ws + QWT_OFF))[idx] = (_Float16)qkv_w[k * 288 + n];
        return;
    }
    idx -= 288 * 96;
    if (idx < 96 * 96) {                          // PWT[n<96][k<96]
        int n = idx / 96, k = idx % 96;
        ((_Float16*)(ws + PWT_OFF))[idx] = (_Float16)proj_w[k * 96 + n];
        return;
    }
    idx -= 96 * 96;
    if (idx < 384 * 96) {                         // F1T[n<384][k<96]
        int n = idx / 96, k = idx % 96;
        ((_Float16*)(ws + F1T_OFF))[idx] = (_Float16)f1w[k * 384 + n];
        return;
    }
    idx -= 384 * 96;
    if (idx < 96 * 384) {                         // F2T[n<96][k<384]
        int n = idx / 384, k = idx % 384;
        ((_Float16*)(ws + F2T_OFF))[idx] = (_Float16)f2w[k * 96 + n];
        return;
    }
}
#define K0W_TOTAL (288*96 + 96*96 + 384*96 + 96*384)

// ---------------------------------------------------------------------------
// k1: per-window gather + LN1 + QKV GEMM (49x96 @ 96x288) via WMMA
//     unconditional padded stores; V stored dd-major (pre-transposed)
// ---------------------------------------------------------------------------
__global__ __launch_bounds__(256) void k1_qkv(
        const float* __restrict__ x, const float* __restrict__ g1,
        const float* __restrict__ bb1, const float* __restrict__ qkv_b,
        char* __restrict__ ws) {
    __shared__ __align__(32) float    xs[N_TOK * C_DIM];   // 18816 B
    __shared__ __align__(32) _Float16 hA[64 * C_DIM];      // 12288 B
    __shared__ float redq[256], redk[256];                 //  2048 B

    int tid = threadIdx.x;
    int w = blockIdx.x;
    int b = w >> 6, widx = w & 63, wh = widx >> 3, wwc = widx & 7;

    // gather shifted window tokens
    for (int idx = tid; idx < N_TOK * C_DIM; idx += 256) {
        int t = idx / C_DIM, c = idx % C_DIM;
        int sh = (wh * WIN + t / WIN + SHIFT_) % HW_;
        int sw = (wwc * WIN + t % WIN + SHIFT_) % HW_;
        xs[idx] = x[((size_t)b * (HW_ * HW_) + sh * HW_ + sw) * C_DIM + c];
    }
    __syncthreads();

    // LayerNorm1 -> f16
    if (tid < N_TOK) {
        float mu = 0.f;
        for (int c = 0; c < C_DIM; ++c) mu += xs[tid * C_DIM + c];
        mu *= (1.0f / C_DIM);
        float var = 0.f;
        for (int c = 0; c < C_DIM; ++c) {
            float d = xs[tid * C_DIM + c] - mu; var += d * d;
        }
        var *= (1.0f / C_DIM);
        float rs = rsqrtf(var + 1e-5f);
        for (int c = 0; c < C_DIM; ++c)
            hA[tid * C_DIM + c] =
                (_Float16)((xs[tid * C_DIM + c] - mu) * rs * g1[c] + bb1[c]);
    }
    for (int idx = tid; idx < (64 - N_TOK) * C_DIM; idx += 256)
        hA[N_TOK * C_DIM + idx] = (_Float16)0.f;
    __syncthreads();

    const _Float16* QWT = (const _Float16*)(ws + QWT_OFF);
    _Float16* Q = (_Float16*)(ws + QKV_OFF);
    _Float16* K = Q + QKV_ELEMS;
    _Float16* V = K + QKV_ELEMS;
    int wv = tid >> 5, lane = tid & 31, lg = lane >> 4, nnn = lane & 15;
    v8f zero = {0.f, 0.f, 0.f, 0.f, 0.f, 0.f, 0.f, 0.f};
    float sq = 0.f, sk = 0.f;

    // 72 output tiles (4 M x 18 N), 9 per wave, 3 K-steps each
    for (int j = 0; j < 9; ++j) {
        int tile = wv * 9 + j;
        int mt = tile & 3, nt = tile >> 2;
        v8f acc = zero;
        for (int ks = 0; ks < 3; ++ks) {
            v16h a  = ldA(hA, C_DIM, mt * 16, ks * 32);
            v16h bf = ldBT(QWT, C_DIM, ks * 32, nt * 16);
            acc = wmma16(a, bf, acc);
        }
        int n = nt * 16 + nnn;
        int g = n / 96, head = (n % 96) / 32, dd = n & 31;   // g, head wave-uniform
        float bias = qkv_b[n];
        size_t hb = (size_t)w * HEADS + head;
        if (g == 2) {            // V: dd-major [w][h][dd][t], t padded to 64
            _Float16* dst = V + hb * (DHEAD * 64) + (size_t)dd * 64;
#pragma unroll
            for (int r = 0; r < 8; ++r) {
                int m = mt * 16 + r + (lg << 3);
                float v = acc[r] + bias;
                dst[m] = (_Float16)v;
            }
        } else {                 // Q/K: token-major [w][h][t(64)][dd]
            _Float16* dst = ((g == 0) ? Q : K) + hb * (64 * DHEAD) + dd;
#pragma unroll
            for (int r = 0; r < 8; ++r) {
                int m = mt * 16 + r + (lg << 3);
                float v = acc[r] + bias;
                dst[(size_t)m * DHEAD] = (_Float16)v;
                float v2 = (m < N_TOK) ? v * v : 0.f;   // guard is pure select
                if (g == 0) sq += v2; else sk += v2;
            }
        }
    }

    // deterministic per-block reduction of sum-of-squares
    redq[tid] = sq; redk[tid] = sk;
    __syncthreads();
    for (int s = 128; s > 0; s >>= 1) {
        if (tid < s) { redq[tid] += redq[tid + s]; redk[tid] += redk[tid + s]; }
        __syncthreads();
    }
    if (tid == 0) {
        float* part = (float*)(ws + PART_OFF);
        part[w * 2 + 0] = redq[0];
        part[w * 2 + 1] = redk[0];
    }
}

// ---------------------------------------------------------------------------
// k1b: deterministic reduction of the 4096 per-window sum-of-squares partials
// ---------------------------------------------------------------------------
__global__ __launch_bounds__(256) void k1b_reduce(char* __restrict__ ws) {
    __shared__ float rq[256], rk[256];
    int tid = threadIdx.x;
    const float* part = (const float*)(ws + PART_OFF);
    float sq = 0.f, sk = 0.f;
    for (int i = tid; i < B_WIN; i += 256) { sq += part[2 * i]; sk += part[2 * i + 1]; }
    rq[tid] = sq; rk[tid] = sk;
    __syncthreads();
    for (int s = 128; s > 0; s >>= 1) {
        if (tid < s) { rq[tid] += rq[tid + s]; rk[tid] += rk[tid + s]; }
        __syncthreads();
    }
    if (tid == 0) {
        float* sums = (float*)(ws + SUMS_OFF);
        sums[0] = rq[0]; sums[1] = rk[0];
    }
}

// ---------------------------------------------------------------------------
// k2: per-window attention (QK^T, scale, bias, mask, softmax, PV, proj) + residual
//     writes Y = x + attn_out into d_out (scattered via inverse shift)
// ---------------------------------------------------------------------------
__global__ __launch_bounds__(256) void k2_attn(
        const float* __restrict__ x, const float* __restrict__ proj_b,
        const float* __restrict__ taup, char* __restrict__ ws,
        float* __restrict__ y) {
    __shared__ __align__(32) char pool[49664];
    _Float16* Qh  = (_Float16*)(pool + 0);       // 64x32 f16 (token-major)
    _Float16* Kh  = (_Float16*)(pool + 4096);    // 64x32 f16 (token-major = N-major)
    _Float16* VhT = (_Float16*)(pool + 8192);    // 32x64 f16 (dd-major = N-major)
    float*    S   = (float*)   (pool + 12288);   // 64x64 f32
    _Float16* P   = (_Float16*)(pool + 28672);   // 64x64 f16
    _Float16* Oc  = (_Float16*)(pool + 36864);   // 64x96 f16 (head concat)
    int*      lab = (int*)     (pool + 49152);   // 49 ints
    int*      offT= (int*)     (pool + 49408);   // 49 ints: scatter offsets

    int tid = threadIdx.x;
    int w = blockIdx.x;
    int b = w >> 6, widx = w & 63, wh = widx >> 3, wwc = widx & 7;
    int wv = tid >> 5, lane = tid & 31, lg = lane >> 4, nnn = lane & 15;

    const float* sums  = (const float*)(ws + SUMS_OFF);
    const float* biasT = (const float*)(ws + BIAS_OFF);
    float inv = 1.0f / (sqrtf(sums[0]) * sqrtf(sums[1]) * fmaxf(taup[0], 0.01f));

    if (tid < N_TOK) {
        int gh = wh * WIN + tid / WIN, gw = wwc * WIN + tid % WIN;
        lab[tid] = region56(gh) * 3 + region56(gw);
        int sh = (gh + SHIFT_) % HW_, sw = (gw + SHIFT_) % HW_;
        offT[tid] = (b * (HW_ * HW_) + sh * HW_ + sw) * C_DIM;
    }

    const _Float16* Qg = (const _Float16*)(ws + QKV_OFF);
    const _Float16* Kg = Qg + QKV_ELEMS;
    const _Float16* Vg = Kg + QKV_ELEMS;
    v8f zero = {0.f, 0.f, 0.f, 0.f, 0.f, 0.f, 0.f, 0.f};

    for (int h = 0; h < HEADS; ++h) {
        // stage Q/K/V: 2048 halfs each = exactly one v8h chunk per thread
        size_t base = ((size_t)w * HEADS + h) * (64 * DHEAD);
        ((v8h*)Qh)[tid]  = ((const v8h*)(Qg + base))[tid];
        ((v8h*)Kh)[tid]  = ((const v8h*)(Kg + base))[tid];
        ((v8h*)VhT)[tid] = ((const v8h*)(Vg + base))[tid];
        __syncthreads();

        // S = Q @ K^T : 16 tiles (4x4), 2 per wave, single K-step (K=32)
        for (int j = 0; j < 2; ++j) {
            int tile = wv * 2 + j;
            int mt = tile >> 2, nt = tile & 3;
            v16h a  = ldA(Qh, DHEAD, mt * 16, 0);
            v16h bf = ldBT(Kh, DHEAD, 0, nt * 16);
            v8f acc = wmma16(a, bf, zero);
            int n = nt * 16 + nnn;
#pragma unroll
            for (int r = 0; r < 8; ++r) {
                int m = mt * 16 + r + (lg << 3);
                float sval;
                if (m < N_TOK && n < N_TOK)
                    sval = acc[r] * inv + biasT[h * 2401 + m * N_TOK + n] +
                           ((lab[m] == lab[n]) ? 0.f : -100.f);
                else
                    sval = -1e30f;
                S[m * 64 + n] = sval;
            }
        }
        __syncthreads();

        // row softmax -> P (f16), pad rows/cols zeroed
        if (tid < 64) {
            if (tid < N_TOK) {
                float mx = -1e30f;
                for (int n = 0; n < N_TOK; ++n) mx = fmaxf(mx, S[tid * 64 + n]);
                float sum = 0.f;
                for (int n = 0; n < N_TOK; ++n) sum += __expf(S[tid * 64 + n] - mx);
                float rsum = 1.0f / sum;
                for (int n = 0; n < N_TOK; ++n)
                    P[tid * 64 + n] = (_Float16)(__expf(S[tid * 64 + n] - mx) * rsum);
                for (int n = N_TOK; n < 64; ++n) P[tid * 64 + n] = (_Float16)0.f;
            } else {
                for (int n = 0; n < 64; ++n) P[tid * 64 + n] = (_Float16)0.f;
            }
        }
        __syncthreads();

        // O = P @ V : 8 tiles (4 M x 2 N), 1 per wave, 2 K-steps (K=64)
        {
            int mt = wv >> 1, nt = wv & 1;
            v8f acc = zero;
            for (int ks = 0; ks < 2; ++ks) {
                v16h a  = ldA(P, 64, mt * 16, ks * 32);
                v16h bf = ldBT(VhT, 64, ks * 32, nt * 16);
                acc = wmma16(a, bf, acc);
            }
            int n = nt * 16 + nnn;
#pragma unroll
            for (int r = 0; r < 8; ++r) {
                int m = mt * 16 + r + (lg << 3);
                Oc[m * C_DIM + h * DHEAD + n] = (_Float16)acc[r];
            }
        }
        __syncthreads();
    }

    // proj: Y = Oc(49x96) @ proj_w(96x96) + proj_b + shortcut, scatter back
    const _Float16* PWT = (const _Float16*)(ws + PWT_OFF);
    for (int j = 0; j < 3; ++j) {
        int tile = wv * 3 + j;
        int mt = tile & 3, nt = tile >> 2;
        v8f acc = zero;
        for (int ks = 0; ks < 3; ++ks) {
            v16h a  = ldA(Oc, C_DIM, mt * 16, ks * 32);
            v16h bf = ldBT(PWT, C_DIM, ks * 32, nt * 16);
            acc = wmma16(a, bf, acc);
        }
        int n = nt * 16 + nnn;
        float pb = proj_b[n];
#pragma unroll
        for (int r = 0; r < 8; ++r) {
            int m = mt * 16 + r + (lg << 3);
            if (m < N_TOK) {
                size_t idx = (size_t)offT[m] + n;
                y[idx] = x[idx] + acc[r] + pb;
            }
        }
    }
}

// ---------------------------------------------------------------------------
// k3: LN2 + MLP (96 -> 384 GELU -> 96) + in-place residual on d_out (= Y)
// one block per 64 tokens (200704 / 64 = 3136 blocks, exact)
// ---------------------------------------------------------------------------
__global__ __launch_bounds__(256) void k3_mlp(
        const float* __restrict__ g2, const float* __restrict__ bb2,
        const float* __restrict__ f1b, const float* __restrict__ f2b,
        char* __restrict__ ws, float* __restrict__ y) {
    __shared__ __align__(32) _Float16 hB[64 * C_DIM];     // 12288 B
    __shared__ __align__(32) _Float16 Hh[64 * HIDDEN];    // 49152 B

    int tid = threadIdx.x;
    size_t base = (size_t)blockIdx.x * 64 * C_DIM;
    int wv = tid >> 5, lane = tid & 31, lg = lane >> 4, nnn = lane & 15;
    v8f zero = {0.f, 0.f, 0.f, 0.f, 0.f, 0.f, 0.f, 0.f};

    if (tid < 64) {
        const float* row = y + base + (size_t)tid * C_DIM;
        float mu = 0.f;
        for (int c = 0; c < C_DIM; ++c) mu += row[c];
        mu *= (1.0f / C_DIM);
        float var = 0.f;
        for (int c = 0; c < C_DIM; ++c) { float d = row[c] - mu; var += d * d; }
        var *= (1.0f / C_DIM);
        float rs = rsqrtf(var + 1e-5f);
        for (int c = 0; c < C_DIM; ++c)
            hB[tid * C_DIM + c] = (_Float16)((row[c] - mu) * rs * g2[c] + bb2[c]);
    }
    __syncthreads();

    const _Float16* F1T = (const _Float16*)(ws + F1T_OFF);
    const _Float16* F2T = (const _Float16*)(ws + F2T_OFF);

    // FC1: 96 tiles (4 M x 24 N), 12 per wave, 3 K-steps
    for (int j = 0; j < 12; ++j) {
        int tile = wv * 12 + j;
        int mt = tile & 3, nt = tile >> 2;
        v8f acc = zero;
        for (int ks = 0; ks < 3; ++ks) {
            v16h a  = ldA(hB, C_DIM, mt * 16, ks * 32);
            v16h bf = ldBT(F1T, C_DIM, ks * 32, nt * 16);
            acc = wmma16(a, bf, acc);
        }
        int n = nt * 16 + nnn;
        float bias = f1b[n];
#pragma unroll
        for (int r = 0; r < 8; ++r) {
            int m = mt * 16 + r + (lg << 3);
            Hh[m * HIDDEN + n] = (_Float16)geluf(acc[r] + bias);
        }
    }
    __syncthreads();

    // FC2: 24 tiles (4 M x 6 N), 3 per wave, 12 K-steps
    for (int j = 0; j < 3; ++j) {
        int tile = wv * 3 + j;
        int mt = tile & 3, nt = tile >> 2;
        v8f acc = zero;
        for (int ks = 0; ks < 12; ++ks) {
            v16h a  = ldA(Hh, HIDDEN, mt * 16, ks * 32);
            v16h bf = ldBT(F2T, HIDDEN, ks * 32, nt * 16);
            acc = wmma16(a, bf, acc);
        }
        int n = nt * 16 + nnn;
        float bias = f2b[n];
#pragma unroll
        for (int r = 0; r < 8; ++r) {
            int m = mt * 16 + r + (lg << 3);
            size_t idx = base + (size_t)m * C_DIM + n;
            y[idx] = y[idx] + acc[r] + bias;   // same-thread RMW, safe
        }
    }
}

// ---------------------------------------------------------------------------
extern "C" void kernel_launch(void* const* d_in, const int* in_sizes, int n_in,
                              void* d_out, int out_size, void* d_ws, size_t ws_size,
                              hipStream_t stream) {
    (void)in_sizes; (void)n_in; (void)out_size; (void)ws_size;
    const float* x      = (const float*)d_in[0];
    const float* n1g    = (const float*)d_in[1];
    const float* n1b    = (const float*)d_in[2];
    const float* qkv_w  = (const float*)d_in[3];
    const float* qkv_b  = (const float*)d_in[4];
    const float* proj_w = (const float*)d_in[5];
    const float* proj_b = (const float*)d_in[6];
    const float* mw1    = (const float*)d_in[7];
    const float* mb1    = (const float*)d_in[8];
    const float* mw2    = (const float*)d_in[9];
    const float* mb2    = (const float*)d_in[10];
    const float* tau    = (const float*)d_in[11];
    const float* n2g    = (const float*)d_in[12];
    const float* n2b    = (const float*)d_in[13];
    const float* f1w    = (const float*)d_in[14];
    const float* f1b    = (const float*)d_in[15];
    const float* f2w    = (const float*)d_in[16];
    const float* f2b    = (const float*)d_in[17];
    char*  ws = (char*)d_ws;
    float* y  = (float*)d_out;

    k0_bias      <<<(N_TOK * N_TOK + 255) / 256, 256, 0, stream>>>(mw1, mb1, mw2, mb2, ws);
    k0w_transpose<<<(K0W_TOTAL + 255) / 256, 256, 0, stream>>>(qkv_w, proj_w, f1w, f2w, ws);
    k1_qkv       <<<B_WIN, 256, 0, stream>>>(x, n1g, n1b, qkv_b, ws);
    k1b_reduce   <<<1, 256, 0, stream>>>(ws);
    k2_attn      <<<B_WIN, 256, 0, stream>>>(x, proj_b, tau, ws, y);
    k3_mlp       <<<3136, 256, 0, stream>>>(n2g, n2b, f1b, f2b, ws, y);
}
